// CrossAttentionFuse_6382321402439
// MI455X (gfx1250) — matrix operand order repaired and checked by reading, not compiled, not run on hardware
//
#include <hip/hip_runtime.h>
#include <hip/hip_bf16.h>

#define B_     4
#define NTOK   2048
#define DMODEL 1024
#define NHEAD  16
#define HD     64

typedef __attribute__((ext_vector_type(16))) __bf16 v16bf;
typedef __attribute__((ext_vector_type(8)))  __bf16 v8bf;
typedef __attribute__((ext_vector_type(8)))  float  v8f;

// Native conversion: lowers to v_cvt_pk_bf16_f32 pairs on gfx1250 (RNE).
__device__ __forceinline__ __bf16 f2bf(float f) { return (__bf16)f; }

// Pack two 8-element (16B) bf16 chunks into a v16bf WMMA operand.
__device__ __forceinline__ v16bf pack2(const __bf16* p0, const __bf16* p1) {
  v8bf a = *(const v8bf*)p0;
  v8bf b = *(const v8bf*)p1;
  v16bf r;
#pragma unroll
  for (int i = 0; i < 8; ++i) { r[i] = a[i]; r[i + 8] = b[i]; }
  return r;
}

// ---------------------------------------------------------------------------
// Stage 1: W[k][n] f32  ->  WT[n][k] bf16   (B-operand wants K contiguous)
// ---------------------------------------------------------------------------
__global__ __launch_bounds__(256) void wtrans_kernel(const float* __restrict__ W,
                                                     __bf16* __restrict__ WT) {
  int idx = blockIdx.x * 256 + threadIdx.x;        // idx = n*1024 + k
  int n = idx >> 10, k = idx & (DMODEL - 1);
  WT[idx] = f2bf(W[(size_t)k * DMODEL + n]);
}

// ---------------------------------------------------------------------------
// Stage 2: projection GEMM.  X[B*N,1024] f32  x  WT[n][k] bf16 -> bf16
// mode 0: Q -> [B,H,N,HD]
// mode 1: K -> [B,H,N,HD], scaled by weight_K[token] * 1/sqrt(HD)
// mode 2: V -> [B,H,HD,N]  (transposed so PV B-tiles are contiguous)
// ---------------------------------------------------------------------------
__global__ __launch_bounds__(256) void proj_kernel(const float* __restrict__ X,
                                                   const __bf16* __restrict__ WT,
                                                   const float* __restrict__ wK,
                                                   __bf16* __restrict__ out,
                                                   int mode) {
  const int lane = threadIdx.x & 31;
  const int wave = threadIdx.x >> 5;
  const int lrow = lane & 15;
  const int ka   = (lane < 16) ? 0 : 8;    // A-operand K base
  const int kb   = (lane < 16) ? 0 : 8;    // B-operand K base
  const int mro  = (lane < 16) ? 0 : 8;    // C/D row offset
  const int m0 = blockIdx.x * 16;
  const int n0 = (blockIdx.y * 8 + wave) * 64;

  v8f acc[4] = {};
  const float* xrow = X + (size_t)(m0 + lrow) * DMODEL;

  for (int kk = 0; kk < DMODEL; kk += 32) {
    const float4 q0 = *(const float4*)(xrow + kk + ka);
    const float4 q1 = *(const float4*)(xrow + kk + ka + 4);
    const float4 q2 = *(const float4*)(xrow + kk + ka + 16);
    const float4 q3 = *(const float4*)(xrow + kk + ka + 20);
    const float fa[16] = {q0.x, q0.y, q0.z, q0.w, q1.x, q1.y, q1.z, q1.w,
                          q2.x, q2.y, q2.z, q2.w, q3.x, q3.y, q3.z, q3.w};
    v16bf a;
#pragma unroll
    for (int i = 0; i < 16; ++i) a[i] = f2bf(fa[i]);
#pragma unroll
    for (int t = 0; t < 4; ++t) {
      const __bf16* wr = WT + (size_t)(n0 + t * 16 + lrow) * DMODEL + kk + kb;
      v16bf b = pack2(wr, wr + 16);
      acc[t] = __builtin_amdgcn_wmma_f32_16x16x32_bf16(false, a, false, b,
                                                       (short)0, acc[t], false, false);
    }
  }

#pragma unroll
  for (int t = 0; t < 4; ++t)
#pragma unroll
    for (int r = 0; r < 8; ++r) {
      int gm = m0 + r + mro;                  // flat token index (b*NTOK+tok)
      int gn = n0 + t * 16 + lrow;            // output channel
      float v = acc[t][r];
      if (mode == 1) v *= wK[gm] * 0.125f;    // weight_K * 1/sqrt(64)
      __bf16 bv = f2bf(v);
      int bb = gm >> 11, tok = gm & (NTOK - 1);
      int h = gn >> 6, d = gn & (HD - 1);
      if (mode == 2)
        out[(((size_t)bb * NHEAD + h) * HD + d) * NTOK + tok] = bv;
      else
        out[(((size_t)bb * NHEAD + h) * NTOK + tok) * HD + d] = bv;
    }
}

// ---------------------------------------------------------------------------
// Stage 3: flash attention.  One wave = 16 q-rows of one (b,h).
// ---------------------------------------------------------------------------
__global__ __launch_bounds__(256) void attn_kernel(const __bf16* __restrict__ Qp,
                                                   const __bf16* __restrict__ Kp,
                                                   const __bf16* __restrict__ Vt,
                                                   __bf16* __restrict__ AO) {
  __shared__ alignas(128) unsigned short Pbuf_s[8][16 * 32];   // 8 KB: per-wave P tile
  const int lane = threadIdx.x & 31;
  const int wave = threadIdx.x >> 5;
  const int lrow = lane & 15;
  const int ka   = (lane < 16) ? 0 : 16;
  const int kb   = (lane < 16) ? 0 : 16;
  const int mro  = (lane < 16) ? 0 : 8;
  const int bh = blockIdx.x;                       // b*NHEAD + h
  const int q0 = blockIdx.y * 128 + wave * 16;

  __bf16* P = (__bf16*)&Pbuf_s[wave][0];

  // Q tile A-operands for head-dim chunks [0,32) and [32,64); resident all loop.
  const __bf16* qrow = Qp + ((size_t)bh * NTOK + q0 + lrow) * HD;
  const v16bf aq0 = pack2(qrow + ka, qrow + ka + 8);
  const v16bf aq1 = pack2(qrow + 32 + ka, qrow + 32 + ka + 8);

  const __bf16* kbase = Kp + (size_t)bh * NTOK * HD;
  const __bf16* vbase = Vt + (size_t)bh * HD * NTOK;

  v8f acc[4] = {};
  float mrow[8], lsum[8];
#pragma unroll
  for (int r = 0; r < 8; ++r) { mrow[r] = -1e30f; lsum[r] = 0.0f; }

  for (int kblk = 0; kblk < NTOK; kblk += 32) {
    // Prefetch next K/V block into cache (global_prefetch_b8) to overlap the
    // serial S -> softmax -> PV chain with the DRAM/L2 stream.
    if (kblk + 32 < NTOK) {
      __builtin_prefetch(kbase + (size_t)(kblk + 32 + lrow) * HD, 0, 1);
      __builtin_prefetch(kbase + (size_t)(kblk + 48 + lrow) * HD, 0, 1);
      __builtin_prefetch(vbase + (size_t)lrow * NTOK + kblk + 32, 0, 1);
    }

    // ---- S = Q K^T for 32 keys (two 16-col D tiles) ----
    v8f s0 = {}, s1 = {};
    {
      const __bf16* kr = kbase + (size_t)(kblk + lrow) * HD;
      v16bf b0 = pack2(kr + kb, kr + kb + 8);            // d in [0,32)
      v16bf b1 = pack2(kr + 32 + kb, kr + 32 + kb + 8);  // d in [32,64)
      s0 = __builtin_amdgcn_wmma_f32_16x16x32_bf16(false, aq0, false, b0, (short)0, s0, false, false);
      s0 = __builtin_amdgcn_wmma_f32_16x16x32_bf16(false, aq1, false, b1, (short)0, s0, false, false);
    }
    {
      const __bf16* kr = kbase + (size_t)(kblk + 16 + lrow) * HD;
      v16bf b0 = pack2(kr + kb, kr + kb + 8);
      v16bf b1 = pack2(kr + 32 + kb, kr + 32 + kb + 8);
      s1 = __builtin_amdgcn_wmma_f32_16x16x32_bf16(false, aq0, false, b0, (short)0, s1, false, false);
      s1 = __builtin_amdgcn_wmma_f32_16x16x32_bf16(false, aq1, false, b1, (short)0, s1, false, false);
    }

    // ---- online softmax over the 32 new scores per row ----
    // Row r lives in lanes [0,16) (M=r) / lanes [16,32) (M=r+8); xor masks <=8
    // reduce each half independently, which is exactly per-row.
#pragma unroll
    for (int r = 0; r < 8; ++r) {
      float mx = fmaxf(s0[r], s1[r]);
#pragma unroll
      for (int m = 8; m >= 1; m >>= 1) mx = fmaxf(mx, __shfl_xor(mx, m, 32));
      float mnew = fmaxf(mrow[r], mx);
      float corr = __expf(mrow[r] - mnew);
      float p0 = __expf(s0[r] - mnew);
      float p1 = __expf(s1[r] - mnew);
      float ps = p0 + p1;
#pragma unroll
      for (int m = 8; m >= 1; m >>= 1) ps += __shfl_xor(ps, m, 32);
      mrow[r] = mnew;
      lsum[r] = lsum[r] * corr + ps;
#pragma unroll
      for (int t = 0; t < 4; ++t) acc[t][r] *= corr;
      // D-layout -> LDS (row-major 16x32 bf16) for A-layout reload
      P[(r + mro) * 32 + lrow]      = f2bf(p0);
      P[(r + mro) * 32 + 16 + lrow] = f2bf(p1);
    }

    // ---- O += P V  (P: 16x32 A-operand, V^T rows contiguous in k) ----
    const __bf16* prow = P + lrow * 32;
    v16bf aP = pack2(prow + ka, prow + ka + 8);
#pragma unroll
    for (int t = 0; t < 4; ++t) {
      const __bf16* vr = vbase + (size_t)(t * 16 + lrow) * NTOK + kblk + kb;
      v16bf bv = pack2(vr, vr + 8);
      acc[t] = __builtin_amdgcn_wmma_f32_16x16x32_bf16(false, aP, false, bv,
                                                       (short)0, acc[t], false, false);
    }
  }

  // ---- normalize and emit attention output [B*N, DMODEL] bf16 ----
  const int b = bh >> 4, h = bh & 15;
#pragma unroll
  for (int t = 0; t < 4; ++t)
#pragma unroll
    for (int r = 0; r < 8; ++r) {
      float o = acc[t][r] / lsum[r];
      size_t row = (size_t)b * NTOK + q0 + r + mro;
      AO[row * DMODEL + h * HD + t * 16 + lrow] = f2bf(o);
    }
}

// ---------------------------------------------------------------------------
// Stage 4: out = AO(bf16) @ WoT(bf16) + bo  -> f32
// ---------------------------------------------------------------------------
__global__ __launch_bounds__(256) void outproj_kernel(const __bf16* __restrict__ AO,
                                                      const __bf16* __restrict__ WoT,
                                                      const float* __restrict__ bo,
                                                      float* __restrict__ out) {
  const int lane = threadIdx.x & 31;
  const int wave = threadIdx.x >> 5;
  const int lrow = lane & 15;
  const int ka   = (lane < 16) ? 0 : 16;
  const int kb   = (lane < 16) ? 0 : 16;
  const int mro  = (lane < 16) ? 0 : 8;
  const int m0 = blockIdx.x * 16;
  const int n0 = (blockIdx.y * 8 + wave) * 64;

  v8f acc[4] = {};
  const __bf16* arow = AO + (size_t)(m0 + lrow) * DMODEL;

  for (int kk = 0; kk < DMODEL; kk += 32) {
    v16bf a = pack2(arow + kk + ka, arow + kk + ka + 8);
#pragma unroll
    for (int t = 0; t < 4; ++t) {
      const __bf16* wr = WoT + (size_t)(n0 + t * 16 + lrow) * DMODEL + kk + kb;
      v16bf b = pack2(wr, wr + 8);
      acc[t] = __builtin_amdgcn_wmma_f32_16x16x32_bf16(false, a, false, b,
                                                       (short)0, acc[t], false, false);
    }
  }

#pragma unroll
  for (int t = 0; t < 4; ++t)
#pragma unroll
    for (int r = 0; r < 8; ++r) {
      int gm = m0 + r + mro;
      int gn = n0 + t * 16 + lrow;
      out[(size_t)gm * DMODEL + gn] = acc[t][r] + bo[gn];
    }
}

// ---------------------------------------------------------------------------
// Host side
// ---------------------------------------------------------------------------
extern "C" void kernel_launch(void* const* d_in, const int* in_sizes, int n_in,
                              void* d_out, int out_size, void* d_ws, size_t ws_size,
                              hipStream_t stream) {
  const float* Qt  = (const float*)d_in[0];
  const float* Kt  = (const float*)d_in[1];
  const float* Vtk = (const float*)d_in[2];
  const float* wK  = (const float*)d_in[3];
  const float* Wq  = (const float*)d_in[4];
  const float* Wk  = (const float*)d_in[5];
  const float* Wv  = (const float*)d_in[6];
  const float* Wo  = (const float*)d_in[7];
  const float* bo  = (const float*)d_in[8];
  float* out = (float*)d_out;

  const size_t wbytes = (size_t)DMODEL * DMODEL * sizeof(__bf16);          // 2 MB
  const size_t pbytes = (size_t)B_ * NHEAD * NTOK * HD * sizeof(__bf16);   // 16 MB
  char* ws = (char*)d_ws;
  __bf16* WqT = (__bf16*)(ws + 0 * wbytes);
  __bf16* WkT = (__bf16*)(ws + 1 * wbytes);
  __bf16* WvT = (__bf16*)(ws + 2 * wbytes);
  __bf16* WoT = (__bf16*)(ws + 3 * wbytes);
  __bf16* Qp  = (__bf16*)(ws + 4 * wbytes);
  __bf16* Kp  = (__bf16*)(ws + 4 * wbytes + 1 * pbytes);
  __bf16* Vt  = (__bf16*)(ws + 4 * wbytes + 2 * pbytes);
  __bf16* AO  = (__bf16*)(ws + 4 * wbytes + 3 * pbytes);

  const int wgrid = (DMODEL * DMODEL) / 256;   // 4096
  wtrans_kernel<<<wgrid, 256, 0, stream>>>(Wq, WqT);
  wtrans_kernel<<<wgrid, 256, 0, stream>>>(Wk, WkT);
  wtrans_kernel<<<wgrid, 256, 0, stream>>>(Wv, WvT);
  wtrans_kernel<<<wgrid, 256, 0, stream>>>(Wo, WoT);

  dim3 ggrid((B_ * NTOK) / 16, DMODEL / (8 * 64));   // (512, 2)
  proj_kernel<<<ggrid, 256, 0, stream>>>(Qt,  WqT, nullptr, Qp, 0);
  proj_kernel<<<ggrid, 256, 0, stream>>>(Kt,  WkT, wK,      Kp, 1);
  proj_kernel<<<ggrid, 256, 0, stream>>>(Vtk, WvT, nullptr, Vt, 2);

  attn_kernel<<<dim3(B_ * NHEAD, NTOK / 128), 256, 0, stream>>>(Qp, Kp, Vt, AO);

  outproj_kernel<<<ggrid, 256, 0, stream>>>(AO, WoT, bo, out);
}
